// LlamaSparseMoeBlock__61409442398449
// MI455X (gfx1250) — matrix operand order repaired and checked by reading, not compiled, not run on hardware
//
#include <hip/hip_runtime.h>
#include <hip/hip_bf16.h>

typedef __bf16 bf16_t;
typedef __attribute__((ext_vector_type(16))) __bf16 v16bf;
typedef __attribute__((ext_vector_type(8)))  float  v8f;

#define TILE_K 32
#define LDS_STRIDE 40   // 32 bf16 + 8 pad = 80 bytes/row -> conflict-free b128 frag loads
#define BM 128

// fp32 -> bf16 (round-to-nearest; ties-away). 2 VALU ops.
static __device__ __forceinline__ unsigned short f2bf(float f) {
  union { float f; unsigned u; } x; x.f = f;
  return (unsigned short)((x.u + 0x8000u) >> 16);
}

// Pack two fp32 -> packed bf16x2 in one dword (v_add x2 + v_perm_b32).
static __device__ __forceinline__ unsigned pack2(float a, float b) {
#if __has_builtin(__builtin_amdgcn_cvt_pk_bf16_f32)
  typedef __attribute__((ext_vector_type(2))) __bf16 v2bf;
  union { v2bf v; unsigned u; } r;
  r.v = __builtin_amdgcn_cvt_pk_bf16_f32(a, b);
  return r.u;
#else
  union { float f; unsigned u; } x, y; x.f = a; y.f = b;
  return __builtin_amdgcn_perm(y.u + 0x8000u, x.u + 0x8000u, 0x07060302u);
#endif
}

static __device__ __forceinline__ v16bf ld_frag2(const bf16_t* p0, const bf16_t* p1) {
  union { v16bf v; uint4 q[2]; } u;
  u.q[0] = *(const uint4*)p0;   // ds_load_b128
  u.q[1] = *(const uint4*)p1;   // ds_load_b128
  return u.v;
}

// ---------------- zero init (out + counts) ----------------
__global__ void moe_zero_kernel(float* __restrict__ out, size_t n, int* __restrict__ counts) {
  size_t i = (size_t)blockIdx.x * blockDim.x + threadIdx.x;
  size_t stride = (size_t)gridDim.x * blockDim.x;
  for (; i < n; i += stride) out[i] = 0.0f;
  if (blockIdx.x == 0 && threadIdx.x < 8) counts[threadIdx.x] = 0;
}

// ---------------- router: logits -> softmax -> top2 -> per-expert lists ----------------
__global__ __launch_bounds__(256)
void moe_router_kernel(const float* __restrict__ hs, const float* __restrict__ gate_w,
                       int* __restrict__ counts, int* __restrict__ tokens,
                       float* __restrict__ wts, int T, int H) {
  __shared__ float logits[8];
  const int t = blockIdx.x;
  const int tid = threadIdx.x;
  const int e = tid >> 5;        // 8 waves, one expert each
  const int lane = tid & 31;
  if (tid < 8) logits[tid] = 0.0f;
  __syncthreads();
  const float* hrow = hs + (size_t)t * H;
  const float* grow = gate_w + (size_t)e * H;
  float p = 0.0f;
  for (int k = lane; k < H; k += 32) p += hrow[k] * grow[k];
  #pragma unroll
  for (int off = 16; off > 0; off >>= 1) p += __shfl_down(p, off, 32);
  if (lane == 0) logits[e] = p;
  __syncthreads();
  if (tid == 0) {
    float m = logits[0];
    #pragma unroll
    for (int i = 1; i < 8; i++) m = fmaxf(m, logits[i]);
    float pr[8]; float s = 0.0f;
    #pragma unroll
    for (int i = 0; i < 8; i++) { pr[i] = __expf(logits[i] - m); s += pr[i]; }
    const float inv = 1.0f / s;
    int i1 = 0;
    #pragma unroll
    for (int i = 1; i < 8; i++) if (pr[i] > pr[i1]) i1 = i;
    int i2 = (i1 == 0) ? 1 : 0;
    #pragma unroll
    for (int i = 0; i < 8; i++) if (i != i1 && pr[i] > pr[i2]) i2 = i;
    int s1 = atomicAdd(&counts[i1], 1);
    tokens[i1 * T + s1] = t; wts[i1 * T + s1] = pr[i1] * inv;  // raw softmax weight (no renorm)
    int s2 = atomicAdd(&counts[i2], 1);
    tokens[i2 * T + s2] = t; wts[i2 * T + s2] = pr[i2] * inv;
  }
}

// ---------------- padded exclusive scan over expert counts ----------------
__global__ void moe_offsets_kernel(const int* __restrict__ counts, int* __restrict__ offs) {
  if (blockIdx.x == 0 && threadIdx.x == 0) {
    int acc = 0;
    for (int e = 0; e < 8; e++) {
      offs[e] = acc;
      acc += (counts[e] + BM - 1) / BM * BM;
    }
  }
}

// ---------------- GEMM1: X @ w1[e]^T, SiLU(gate)*up -> h (bf16) ----------------
__global__ __launch_bounds__(256)
void moe_gemm1_kernel(const float* __restrict__ hs, const float* __restrict__ w1,
                      const int* __restrict__ counts, const int* __restrict__ offs,
                      const int* __restrict__ tokens, bf16_t* __restrict__ h_ws,
                      int T, int H, int I) {
  const int e = blockIdx.z;
  const int cnt = counts[e];
  const int tileM = blockIdx.y * BM;
  if (tileM >= cnt) return;                     // block-uniform exit (EXEC stays all-ones)
  const int nbase = blockIdx.x * 64;            // intermediate columns [nbase, nbase+64)

  __shared__ bf16_t Alds[BM * LDS_STRIDE];      // 128 token rows x 32 K (bf16)
  __shared__ bf16_t Blds[BM * LDS_STRIDE];      // rows 0..63 gate, 64..127 up

  const int tid = threadIdx.x;
  const float* w1e = w1 + (size_t)e * 2u * (size_t)I * H;

  const int col4 = tid & 7;                     // float4 column within 32-K row
  const int arow = tid >> 3;                    // 0..31
  const float* arow_ptr[4];
  const float* brow_ptr[4];
  #pragma unroll
  for (int r = 0; r < 4; r++) {
    int slot = tileM + arow + 32 * r;
    int cs = slot < cnt ? slot : cnt - 1;       // clamp pad rows to a valid token
    int tok = tokens[e * T + cs];
    arow_ptr[r] = hs + (size_t)tok * H + col4 * 4;
    int row = arow + 32 * r;
    int w1row = (row < 64) ? (nbase + row) : (I + nbase + row - 64);
    brow_ptr[r] = w1e + (size_t)w1row * H + col4 * 4;
  }

  const int wave = tid >> 5, lane = tid & 31;
  const int wm = wave & 3, wn = wave >> 2;      // 4 M-groups x 2 N-groups
  const int lm = lane & 15;
  const int khalfA = (lane < 16) ? 0 : 8;
  const int kbB    = (lane < 16) ? 0 : 8;

  v8f acc[2][2][2] = {};                        // [mTile][nTile][gate/up]

  for (int k0 = 0; k0 < H; k0 += TILE_K) {
    #pragma unroll
    for (int r = 0; r < 4; r++) {
      float4 a = *(const float4*)(arow_ptr[r] + k0);
      uint2 pa; pa.x = pack2(a.x, a.y); pa.y = pack2(a.z, a.w);
      *(uint2*)&Alds[(arow + 32 * r) * LDS_STRIDE + col4 * 4] = pa;
      float4 b = *(const float4*)(brow_ptr[r] + k0);
      uint2 pb; pb.x = pack2(b.x, b.y); pb.y = pack2(b.z, b.w);
      *(uint2*)&Blds[(arow + 32 * r) * LDS_STRIDE + col4 * 4] = pb;
      // speculative prefetch (dropped on invalid addr): branch-free, 2 tiles ahead
      __builtin_prefetch(brow_ptr[r] + k0 + 2 * TILE_K, 0, 1);
    }
    __syncthreads();

    v16bf af[2], bg[2], bu[2];
    #pragma unroll
    for (int mt = 0; mt < 2; mt++) {
      int row = wm * 32 + mt * 16 + lm;
      af[mt] = ld_frag2(&Alds[row * LDS_STRIDE + khalfA],
                        &Alds[row * LDS_STRIDE + khalfA + 16]);
    }
    #pragma unroll
    for (int nt = 0; nt < 2; nt++) {
      int gn = wn * 32 + nt * 16 + lm;
      bg[nt] = ld_frag2(&Blds[gn * LDS_STRIDE + kbB], &Blds[gn * LDS_STRIDE + kbB + 16]);
      int un = 64 + gn;
      bu[nt] = ld_frag2(&Blds[un * LDS_STRIDE + kbB], &Blds[un * LDS_STRIDE + kbB + 16]);
    }
    #pragma unroll
    for (int mt = 0; mt < 2; mt++)
      #pragma unroll
      for (int nt = 0; nt < 2; nt++) {
        acc[mt][nt][0] = __builtin_amdgcn_wmma_f32_16x16x32_bf16(
            false, af[mt], false, bg[nt], (short)0, acc[mt][nt][0], false, false);
        acc[mt][nt][1] = __builtin_amdgcn_wmma_f32_16x16x32_bf16(
            false, af[mt], false, bu[nt], (short)0, acc[mt][nt][1], false, false);
      }
    __syncthreads();
  }

  // SiLU(g)*u -> bf16 h  (write all 128 rows incl. pads; pads skipped in pass 2)
  unsigned short* hbase = (unsigned short*)(h_ws + (size_t)offs[e] * I);
  const int mlane = (lane < 16) ? 0 : 8;
  #pragma unroll
  for (int mt = 0; mt < 2; mt++)
    #pragma unroll
    for (int nt = 0; nt < 2; nt++) {
      int col = nbase + wn * 32 + nt * 16 + lm;
      #pragma unroll
      for (int i = 0; i < 8; i++) {
        float g = acc[mt][nt][0][i];
        float u = acc[mt][nt][1][i];
        // sigmoid via hardware v_rcp_f32 (~1 ulp) instead of IEEE div sequence
        float sig = __builtin_amdgcn_rcpf(1.0f + __expf(-g));
        float hv = g * sig * u;
        int slot = tileM + wm * 32 + mt * 16 + mlane + i;
        hbase[(size_t)slot * I + col] = f2bf(hv);
      }
    }
}

// ---------------- GEMM2: h @ w2[e]^T, scale, scatter-add into out ----------------
__global__ __launch_bounds__(256)
void moe_gemm2_kernel(const bf16_t* __restrict__ h_ws, const float* __restrict__ w2,
                      const int* __restrict__ counts, const int* __restrict__ offs,
                      const int* __restrict__ tokens, const float* __restrict__ wts,
                      float* __restrict__ out, int T, int H, int I) {
  const int e = blockIdx.z;
  const int cnt = counts[e];
  const int tileM = blockIdx.y * BM;
  if (tileM >= cnt) return;
  const int nbase = blockIdx.x * 128;           // H columns

  __shared__ bf16_t Alds[BM * LDS_STRIDE];
  __shared__ bf16_t Blds[BM * LDS_STRIDE];

  const int tid = threadIdx.x;
  const float* w2e = w2 + (size_t)e * (size_t)H * I;
  const bf16_t* hbase = h_ws + (size_t)offs[e] * I;

  const int ac = tid & 3, ar = tid >> 2;        // A: 8-bf16 chunks, rows 0..63 (x2)
  const int col4 = tid & 7, br = tid >> 3;      // B: float4 cols, rows 0..31 (x4)

  const int wave = tid >> 5, lane = tid & 31;
  const int wm = wave & 3, wn = wave >> 2;      // wave tile: 32 rows x 64 cols
  const int lm = lane & 15;
  const int khalfA = (lane < 16) ? 0 : 8;
  const int kbB    = (lane < 16) ? 0 : 8;

  v8f acc[2][4] = {};

  for (int k0 = 0; k0 < I; k0 += TILE_K) {
    #pragma unroll
    for (int r = 0; r < 2; r++) {
      int row = ar + 64 * r;
      uint4 av = *(const uint4*)(hbase + (size_t)(tileM + row) * I + k0 + ac * 8);
      *(uint4*)&Alds[row * LDS_STRIDE + ac * 8] = av;
    }
    #pragma unroll
    for (int r = 0; r < 4; r++) {
      int row = br + 32 * r;
      const float* bp = w2e + (size_t)(nbase + row) * I + k0 + col4 * 4;
      float4 b = *(const float4*)bp;
      uint2 pb; pb.x = pack2(b.x, b.y); pb.y = pack2(b.z, b.w);
      *(uint2*)&Blds[row * LDS_STRIDE + col4 * 4] = pb;
      __builtin_prefetch(bp + 2 * TILE_K, 0, 1);   // speculative, branch-free
    }
    __syncthreads();

    v16bf af[2], bfrag[4];
    #pragma unroll
    for (int mt = 0; mt < 2; mt++) {
      int row = wm * 32 + mt * 16 + lm;
      af[mt] = ld_frag2(&Alds[row * LDS_STRIDE + khalfA],
                        &Alds[row * LDS_STRIDE + khalfA + 16]);
    }
    #pragma unroll
    for (int nt = 0; nt < 4; nt++) {
      int n = wn * 64 + nt * 16 + lm;
      bfrag[nt] = ld_frag2(&Blds[n * LDS_STRIDE + kbB], &Blds[n * LDS_STRIDE + kbB + 16]);
    }
    #pragma unroll
    for (int mt = 0; mt < 2; mt++)
      #pragma unroll
      for (int nt = 0; nt < 4; nt++)
        acc[mt][nt] = __builtin_amdgcn_wmma_f32_16x16x32_bf16(
            false, af[mt], false, bfrag[nt], (short)0, acc[mt][nt], false, false);
    __syncthreads();
  }

  const int mlane = (lane < 16) ? 0 : 8;
  #pragma unroll
  for (int mt = 0; mt < 2; mt++) {
    #pragma unroll
    for (int i = 0; i < 8; i++) {
      int slot = tileM + wm * 32 + mt * 16 + mlane + i;
      if (slot < cnt) {
        int tok = tokens[e * T + slot];
        float sc = wts[e * T + slot];
        float* orow = out + (size_t)tok * H;
        #pragma unroll
        for (int nt = 0; nt < 4; nt++) {
          int col = nbase + wn * 64 + nt * 16 + lm;
          unsafeAtomicAdd(&orow[col], sc * acc[mt][nt][i]);  // global_atomic_add_f32
        }
      }
    }
  }
}

extern "C" void kernel_launch(void* const* d_in, const int* in_sizes, int n_in,
                              void* d_out, int out_size, void* d_ws, size_t ws_size,
                              hipStream_t stream) {
  const float* hs     = (const float*)d_in[0];   // [T,H]
  const float* gate_w = (const float*)d_in[1];   // [E,H]
  const float* w1     = (const float*)d_in[2];   // [E,2I,H]
  const float* w2     = (const float*)d_in[3];   // [E,H,I]
  float* out = (float*)d_out;
  const int T = 2048, H = 2048, I = 5632, E = 8;

  // workspace layout: counts | offs | tokens[E*T] | wts[E*T] | h (bf16, padded-compact)
  char* ws = (char*)d_ws;
  int*    counts = (int*)ws;
  int*    offs   = (int*)(ws + 256);
  int*    tokens = (int*)(ws + 512);
  float*  wts    = (float*)(ws + 512 + (size_t)E * T * 4);
  bf16_t* h_ws   = (bf16_t*)(ws + 512 + 2 * (size_t)E * T * 4);
  // h capacity needed: (T*topk + E*(BM-1)) rows * I * 2B  ~= 58 MB

  moe_zero_kernel<<<2048, 256, 0, stream>>>(out, (size_t)T * H, counts);
  moe_router_kernel<<<T, 256, 0, stream>>>(hs, gate_w, counts, tokens, wts, T, H);
  moe_offsets_kernel<<<1, 32, 0, stream>>>(counts, offs);
  moe_gemm1_kernel<<<dim3(I / 64, T / BM, E), 256, 0, stream>>>(
      hs, w1, counts, offs, tokens, h_ws, T, H, I);
  moe_gemm2_kernel<<<dim3(H / 128, T / BM, E), 256, 0, stream>>>(
      h_ws, w2, counts, offs, tokens, wts, out, T, H, I);
}
